// VanillaRNN_59382217835037
// MI455X (gfx1250) — matrix operand order repaired and checked by reading, not compiled, not run on hardware
//
#include <hip/hip_runtime.h>
#include <math.h>

#define B_  512
#define T_  256
#define D_  256
#define H_  1024
#define O_  128

typedef __bf16 bf16;
typedef __attribute__((ext_vector_type(16))) __bf16 v16bf;
typedef __attribute__((ext_vector_type(8)))  __bf16 v8bf;
typedef __attribute__((ext_vector_type(8)))  float  v8f;

#define WMMA_BF16(a, b, c) \
  __builtin_amdgcn_wmma_f32_16x16x32_bf16(false, (a), false, (b), (short)0, (c), false, false)

// ---------------------------------------------------------------------------
// Fragment loaders (wave32). ISA layouts (CDNA5 ISA §7.12.2):
//  A 16x32 bf16: lane m=lane&15; half=lane>>4: K = kb..kb+7 (VGPR0-3) and
//                kb+16..kb+23 (VGPR4-7), kb = kc + half*8.
//  B 32x16 bf16 (stored as weight rows [N][K]): lane n=lane&15; half selects
//                K = kb..kb+7 and kb+16..kb+23, kb = kc + half*8 (same as A).
// ---------------------------------------------------------------------------
__device__ __forceinline__ v16bf fragA_bf16(const bf16* __restrict__ row, int kc, int half) {
  const bf16* p = row + kc + half * 8;
  v8bf lo = *(const v8bf*)(p);
  v8bf hi = *(const v8bf*)(p + 16);
  v16bf a;
#pragma unroll
  for (int i = 0; i < 8; ++i) { a[i] = lo[i]; a[i + 8] = hi[i]; }
  return a;
}

__device__ __forceinline__ v16bf fragA_f32(const float* __restrict__ row, int kc, int half) {
  const float* p = row + kc + half * 8;
  v8f lo = *(const v8f*)(p);
  v8f hi = *(const v8f*)(p + 16);
  v16bf a;
#pragma unroll
  for (int i = 0; i < 8; ++i) { a[i] = (bf16)lo[i]; a[i + 8] = (bf16)hi[i]; }
  return a;
}

__device__ __forceinline__ v16bf fragB_bf16(const bf16* __restrict__ row, int kc, int half) {
  const bf16* p = row + kc + half * 8;
  v8bf lo = *(const v8bf*)(p);
  v8bf hi = *(const v8bf*)(p + 16);
  v16bf b;
#pragma unroll
  for (int i = 0; i < 8; ++i) { b[i] = lo[i]; b[i + 8] = hi[i]; }
  return b;
}

// ---------------------------------------------------------------------------
// One recurrence step: hout = tanh(x_t @ Wx^T + hin @ Wh^T + bias)
// Fused K = D_ (from f32 x, converted in-register) + H_ (from bf16 h).
// Latency-optimized geometry for the sequential scan: 64-thread blocks
// (2 waves), wave tile 32x32 (2x2 accums), block tile 32x64.
// Grid = (B_/32, H_/64) = (16, 16) = 256 workgroups -> spreads each
// dependent step across up to 256 WGPs (all operands L2-resident).
// ---------------------------------------------------------------------------
__global__ __launch_bounds__(64) void rnn_step_kernel(
    const float* __restrict__ xt,     // &x[0][t][0], batch row stride T_*D_
    const bf16*  __restrict__ Wx,     // [H_][D_] bf16
    const bf16*  __restrict__ Wh,     // [H_][H_] bf16
    const float* __restrict__ bias,   // [H_] = Wx_b + Wh_b
    const bf16*  __restrict__ hin,    // [B_][H_] bf16
    bf16*        __restrict__ hout)   // [B_][H_] bf16
{
  const int lane = threadIdx.x & 31;
  const int wave = threadIdx.x >> 5;
  const int half = lane >> 4;
  const int l16  = lane & 15;
  const int m0 = blockIdx.x * 32;
  const int n0 = blockIdx.y * 64 + wave * 32;

  v8f c00 = {}, c01 = {}, c10 = {}, c11 = {};

  // ---- x_t part (K = D_), f32 source converted in-register to bf16 ----
  const float* xa0 = xt + (size_t)(m0 + l16)      * (size_t)(T_ * D_);
  const float* xa1 = xt + (size_t)(m0 + 16 + l16) * (size_t)(T_ * D_);
  const bf16*  wx0 = Wx + (size_t)(n0 + l16)      * D_;
  const bf16*  wx1 = Wx + (size_t)(n0 + 16 + l16) * D_;
#pragma unroll 2
  for (int kc = 0; kc < D_; kc += 32) {
    v16bf a0 = fragA_f32(xa0, kc, half);
    v16bf a1 = fragA_f32(xa1, kc, half);
    v16bf b0 = fragB_bf16(wx0, kc, half);
    v16bf b1 = fragB_bf16(wx1, kc, half);
    c00 = WMMA_BF16(a0, b0, c00);
    c01 = WMMA_BF16(a0, b1, c01);
    c10 = WMMA_BF16(a1, b0, c10);
    c11 = WMMA_BF16(a1, b1, c11);
  }

  // ---- h part (K = H_), bf16 source ----
  const bf16* ha0 = hin + (size_t)(m0 + l16)      * H_;
  const bf16* ha1 = hin + (size_t)(m0 + 16 + l16) * H_;
  const bf16* wh0 = Wh  + (size_t)(n0 + l16)      * H_;
  const bf16* wh1 = Wh  + (size_t)(n0 + 16 + l16) * H_;
#pragma unroll 2
  for (int kc = 0; kc < H_; kc += 32) {
    __builtin_prefetch(wh0 + kc + 128, 0, 1);   // global_prefetch_b8: warm B path
    v16bf a0 = fragA_bf16(ha0, kc, half);
    v16bf a1 = fragA_bf16(ha1, kc, half);
    v16bf b0 = fragB_bf16(wh0, kc, half);
    v16bf b1 = fragB_bf16(wh1, kc, half);
    c00 = WMMA_BF16(a0, b0, c00);
    c01 = WMMA_BF16(a0, b1, c01);
    c10 = WMMA_BF16(a1, b0, c10);
    c11 = WMMA_BF16(a1, b1, c11);
  }

  // ---- epilogue: + bias, tanh, store bf16 ----
  // C/D layout: VGPR r holds M = r + 8*half; N = lane&15.
  const float bn0 = bias[n0 + l16];
  const float bn1 = bias[n0 + 16 + l16];
  const int rbase = half * 8;
#pragma unroll
  for (int r = 0; r < 8; ++r) {
    const size_t row0 = (size_t)(m0 + rbase + r) * H_;
    const size_t row1 = (size_t)(m0 + 16 + rbase + r) * H_;
    hout[row0 + n0 + l16]      = (bf16)tanhf(c00[r] + bn0);
    hout[row0 + n0 + 16 + l16] = (bf16)tanhf(c01[r] + bn1);
    hout[row1 + n0 + l16]      = (bf16)tanhf(c10[r] + bn0);
    hout[row1 + n0 + 16 + l16] = (bf16)tanhf(c11[r] + bn1);
  }
}

// ---------------------------------------------------------------------------
// y = h_final @ Wy^T + Wy_b   (M=512, N=128, K=1024) -> f32 scratch buffer
// Grid = (B_/64, 1); 8 waves cover block tile 64x128 (whole N).
// ---------------------------------------------------------------------------
__global__ __launch_bounds__(256) void out_gemm_kernel(
    const bf16*  __restrict__ hfin,   // [B_][H_] bf16
    const bf16*  __restrict__ Wy,     // [O_][H_] bf16
    const float* __restrict__ yb,     // [O_]
    float*       __restrict__ y)      // [B_][O_]
{
  const int lane = threadIdx.x & 31;
  const int wave = threadIdx.x >> 5;
  const int half = lane >> 4;
  const int l16  = lane & 15;
  const int m0 = blockIdx.x * 64 + (wave >> 2) * 32;
  const int n0 = (wave & 3) * 32;

  v8f c00 = {}, c01 = {}, c10 = {}, c11 = {};

  const bf16* ha0 = hfin + (size_t)(m0 + l16)      * H_;
  const bf16* ha1 = hfin + (size_t)(m0 + 16 + l16) * H_;
  const bf16* wy0 = Wy   + (size_t)(n0 + l16)      * H_;
  const bf16* wy1 = Wy   + (size_t)(n0 + 16 + l16) * H_;
#pragma unroll 2
  for (int kc = 0; kc < H_; kc += 32) {
    v16bf a0 = fragA_bf16(ha0, kc, half);
    v16bf a1 = fragA_bf16(ha1, kc, half);
    v16bf b0 = fragB_bf16(wy0, kc, half);
    v16bf b1 = fragB_bf16(wy1, kc, half);
    c00 = WMMA_BF16(a0, b0, c00);
    c01 = WMMA_BF16(a0, b1, c01);
    c10 = WMMA_BF16(a1, b0, c10);
    c11 = WMMA_BF16(a1, b1, c11);
  }

  const float bn0 = yb[n0 + l16];
  const float bn1 = yb[n0 + 16 + l16];
  const int rbase = half * 8;
#pragma unroll
  for (int r = 0; r < 8; ++r) {
    const size_t row0 = (size_t)(m0 + rbase + r) * O_;
    const size_t row1 = (size_t)(m0 + 16 + rbase + r) * O_;
    y[row0 + n0 + l16]      = c00[r] + bn0;
    y[row0 + n0 + 16 + l16] = c01[r] + bn1;
    y[row1 + n0 + l16]      = c10[r] + bn0;
    y[row1 + n0 + 16 + l16] = c11[r] + bn1;
  }
}

// ---------------------------------------------------------------------------
// Row softmax over O_=128; one wave32 per row, 4 elements/lane.
// ---------------------------------------------------------------------------
__global__ __launch_bounds__(32) void softmax_kernel(
    const float* __restrict__ y, float* __restrict__ out)
{
  const int b = blockIdx.x;
  const int lane = threadIdx.x;
  float v[4];
  float m = -INFINITY;
#pragma unroll
  for (int j = 0; j < 4; ++j) {
    v[j] = y[(size_t)b * O_ + lane + 32 * j];
    m = fmaxf(m, v[j]);
  }
#pragma unroll
  for (int off = 16; off > 0; off >>= 1) m = fmaxf(m, __shfl_xor(m, off));
  float s = 0.f;
#pragma unroll
  for (int j = 0; j < 4; ++j) { v[j] = expf(v[j] - m); s += v[j]; }
#pragma unroll
  for (int off = 16; off > 0; off >>= 1) s += __shfl_xor(s, off);
  const float inv = 1.f / s;
#pragma unroll
  for (int j = 0; j < 4; ++j) out[(size_t)b * O_ + lane + 32 * j] = v[j] * inv;
}

// ---------------------------------------------------------------------------
// Prep kernels
// ---------------------------------------------------------------------------
__global__ void cvt_f32_bf16_kernel(const float* __restrict__ in, bf16* __restrict__ o, int n) {
  int i = blockIdx.x * blockDim.x + threadIdx.x;
  if (i < n) o[i] = (bf16)in[i];
}
__global__ void bias_sum_kernel(const float* __restrict__ a, const float* __restrict__ b,
                                float* __restrict__ o, int n) {
  int i = blockIdx.x * blockDim.x + threadIdx.x;
  if (i < n) o[i] = a[i] + b[i];
}
__global__ void zero_bf16_kernel(bf16* __restrict__ p, int n) {
  int i = blockIdx.x * blockDim.x + threadIdx.x;
  if (i < n) p[i] = (bf16)0.f;
}

// ---------------------------------------------------------------------------
extern "C" void kernel_launch(void* const* d_in, const int* in_sizes, int n_in,
                              void* d_out, int out_size, void* d_ws, size_t ws_size,
                              hipStream_t stream) {
  (void)in_sizes; (void)n_in; (void)out_size; (void)ws_size;
  const float* x    = (const float*)d_in[0];   // [B_,T_,D_]
  const float* Wx_w = (const float*)d_in[1];   // [H_,D_]
  const float* Wx_b = (const float*)d_in[2];   // [H_]
  const float* Wh_w = (const float*)d_in[3];   // [H_,H_]
  const float* Wh_b = (const float*)d_in[4];   // [H_]
  const float* Wy_w = (const float*)d_in[5];   // [O_,H_]
  const float* Wy_b = (const float*)d_in[6];   // [O_]
  float* out = (float*)d_out;                  // [B_,O_]

  // Workspace layout (~5.2 MB)
  char* ws = (char*)d_ws;
  bf16*  Wxb  = (bf16*)ws;  ws += (size_t)H_ * D_ * sizeof(bf16);
  bf16*  Whb  = (bf16*)ws;  ws += (size_t)H_ * H_ * sizeof(bf16);
  bf16*  Wyb  = (bf16*)ws;  ws += (size_t)O_ * H_ * sizeof(bf16);
  float* bias = (float*)ws; ws += (size_t)H_ * sizeof(float);
  bf16*  h0   = (bf16*)ws;  ws += (size_t)B_ * H_ * sizeof(bf16);
  bf16*  h1   = (bf16*)ws;  ws += (size_t)B_ * H_ * sizeof(bf16);
  float* ybuf = (float*)ws; ws += (size_t)B_ * O_ * sizeof(float);

  // Prep: weight conversion, fused bias, zero h0
  cvt_f32_bf16_kernel<<<(H_ * D_ + 255) / 256, 256, 0, stream>>>(Wx_w, Wxb, H_ * D_);
  cvt_f32_bf16_kernel<<<(H_ * H_ + 255) / 256, 256, 0, stream>>>(Wh_w, Whb, H_ * H_);
  cvt_f32_bf16_kernel<<<(O_ * H_ + 255) / 256, 256, 0, stream>>>(Wy_w, Wyb, O_ * H_);
  bias_sum_kernel<<<(H_ + 255) / 256, 256, 0, stream>>>(Wx_b, Wh_b, bias, H_);
  zero_bf16_kernel<<<(B_ * H_ + 255) / 256, 256, 0, stream>>>(h0, B_ * H_);

  // Sequential recurrence: 256 launches, h double-buffered.
  bf16* bufs[2] = { h0, h1 };
  dim3 sgrid(B_ / 32, H_ / 64);   // 256 WGs x 2 waves: max WGP spread per step
  for (int t = 0; t < T_; ++t) {
    rnn_step_kernel<<<sgrid, 64, 0, stream>>>(
        x + (size_t)t * D_, Wxb, Whb, bias, bufs[t & 1], bufs[(t + 1) & 1]);
  }
  // After 256 steps final h is in bufs[0].
  out_gemm_kernel<<<dim3(B_ / 64, 1), 256, 0, stream>>>(bufs[0], Wyb, Wy_b, ybuf);
  softmax_kernel<<<B_, 32, 0, stream>>>(ybuf, out);
}
